// SelectiveSSM_43782896615788
// MI455X (gfx1250) — matrix operand order repaired and checked by reading, not compiled, not run on hardware
//
#include <hip/hip_runtime.h>
#include <hip/hip_bf16.h>
#include <math.h>

// ---------------------------------------------------------------------------
// Problem sizes (fixed by reference): B=8, L=2048, D=1024, S=256
// ---------------------------------------------------------------------------
#define BSZ   8
#define LEN   2048
#define DIM   1024
#define SDIM  256
#define ROWS  (BSZ * LEN)   // 16384

typedef __attribute__((ext_vector_type(16))) __bf16 v16bf;
typedef __attribute__((ext_vector_type(8)))  __bf16 v8bf;
typedef __attribute__((ext_vector_type(8)))  float  v8f;
typedef __attribute__((ext_vector_type(4)))  unsigned int v4u;
typedef __attribute__((ext_vector_type(8)))  int v8i;
typedef __attribute__((ext_vector_type(4)))  int v4i;

union AFragU { v16bf v; v8bf h[2]; };

// ---------------------------------------------------------------------------
// WMMA fragment helpers (contiguous k order: k = half*16 + e, wave32)
// ---------------------------------------------------------------------------

// A-matrix 16x32 bf16 fragment. Lane (l<16) holds row M=l, K = half*16 + {0..15}
__device__ __forceinline__ v16bf load_a_bf16(const __bf16* __restrict__ A, int lda,
                                             int row, int k0, int lane) {
  const int half = lane >> 4;
  const __bf16* p = A + (size_t)row * lda + k0 + half * 16;
  AFragU f;
  f.h[0] = *(const v8bf*)(p);        // K = k0 + half*16 + {0..7}
  f.h[1] = *(const v8bf*)(p + 8);    // K = k0 + half*16 + 8 + {0..7}
  return f.v;
}

__device__ __forceinline__ v16bf zero_a16() {
  v16bf z;
#pragma unroll
  for (int i = 0; i < 16; ++i) z[i] = (__bf16)0.0f;
  return z;
}

// B-matrix 32x16 bf16 fragment taken from B-TRANSPOSED storage BT[N][K]:
// lane (half,col) holds B[k0+half*16+e][n] = BT[n][k0+half*16+e], e=0..15
// => one contiguous 32-byte run: two b128 loads, no packing.
__device__ __forceinline__ v16bf load_bt_bf16(const __bf16* __restrict__ BT, int ldk,
                                              int k0, int n, int lane) {
  const int half = lane >> 4;
  const __bf16* p = BT + (size_t)n * ldk + k0 + half * 16;
  AFragU f;
  f.h[0] = *(const v8bf*)(p);
  f.h[1] = *(const v8bf*)(p + 8);
  return f.v;
}

__device__ __forceinline__ v8f wmma_bf16(v16bf a, v16bf b, v8f c) {
  // (neg_a, A, neg_b, B, c_mod, C, reuse_a, reuse_b)
  return __builtin_amdgcn_wmma_f32_16x16x32_bf16(false, a, false, b, (short)0, c,
                                                 false, false);
}

// ---------------------------------------------------------------------------
// Kernel 0: fp32 -> bf16 conversions. All GEMM B-operands stored TRANSPOSED:
//   BmT[s][d], CT[d][s], AT[n][k], wtT[kk][dout][din] (permuted from native conv_w[dout][din][kk])
// ---------------------------------------------------------------------------
__global__ void k_convert(const float* __restrict__ x, const float* __restrict__ A,
                          const float* __restrict__ Bm, const float* __restrict__ C,
                          const float* __restrict__ convw,
                          __bf16* __restrict__ x_bf, __bf16* __restrict__ AT_bf,
                          __bf16* __restrict__ BmT_bf, __bf16* __restrict__ CT_bf,
                          __bf16* __restrict__ wtT_bf) {
  const size_t i0 = (size_t)blockIdx.x * blockDim.x + threadIdx.x;
  const size_t stride = (size_t)gridDim.x * blockDim.x;
  for (size_t j = i0; j < (size_t)ROWS * DIM; j += stride) x_bf[j] = (__bf16)x[j];
  for (size_t j = i0; j < (size_t)SDIM * SDIM; j += stride) {
    const int n = (int)(j >> 8), k = (int)(j & (SDIM - 1));
    AT_bf[j] = (__bf16)A[(size_t)k * SDIM + n];
  }
  for (size_t j = i0; j < (size_t)SDIM * DIM; j += stride) {
    const int s = (int)(j >> 10), d = (int)(j & (DIM - 1));
    BmT_bf[j] = (__bf16)Bm[(size_t)d * SDIM + s];          // BmT[s][d]
  }
  for (size_t j = i0; j < (size_t)DIM * SDIM; j += stride) {
    const int d = (int)(j >> 8), s = (int)(j & (SDIM - 1));
    CT_bf[j] = (__bf16)C[(size_t)s * DIM + d];             // CT[d][s]
  }
  for (size_t j = i0; j < (size_t)3 * DIM * DIM; j += stride) {
    const int kk   = (int)(j >> 20);
    const int dout = (int)((j >> 10) & (DIM - 1));
    const int din  = (int)(j & (DIM - 1));
    wtT_bf[j] = (__bf16)convw[((size_t)dout * DIM + din) * 3 + kk];  // wtT[kk][dout][din]
  }
}

// ---------------------------------------------------------------------------
// Kernel 1: xB = x_bf (16384x1024) @ Bm (1024x256) -> f32, B from BmT[s][d]
// block = 256 thr = 8 waves (4Mx2N of 32x32 wave tiles) => 128x64 block tile
// ---------------------------------------------------------------------------
__global__ void __launch_bounds__(256) k_gemm_xb(const __bf16* __restrict__ Xb,
                                                 const __bf16* __restrict__ BmT,
                                                 float* __restrict__ xB) {
  const int lane = threadIdx.x & 31;
  const int wave = threadIdx.x >> 5;
  const int m0 = blockIdx.x * 128 + (wave >> 1) * 32;
  const int n0 = blockIdx.y * 64 + (wave & 1) * 32;
  const int half = lane >> 4, col = lane & 15;

  v8f acc[2][2] = {{}, {}};
  for (int k0 = 0; k0 < DIM; k0 += 32) {
    __builtin_prefetch(Xb + (size_t)(m0 + col) * DIM + k0 + 128, 0, 0);
    v16bf a0 = load_a_bf16(Xb, DIM, m0 + col, k0, lane);
    v16bf a1 = load_a_bf16(Xb, DIM, m0 + 16 + col, k0, lane);
    v16bf b0 = load_bt_bf16(BmT, DIM, k0, n0 + col, lane);
    v16bf b1 = load_bt_bf16(BmT, DIM, k0, n0 + 16 + col, lane);
    acc[0][0] = wmma_bf16(a0, b0, acc[0][0]);
    acc[0][1] = wmma_bf16(a0, b1, acc[0][1]);
    acc[1][0] = wmma_bf16(a1, b0, acc[1][0]);
    acc[1][1] = wmma_bf16(a1, b1, acc[1][1]);
  }
#pragma unroll
  for (int i = 0; i < 2; ++i)
#pragma unroll
    for (int j = 0; j < 2; ++j)
#pragma unroll
      for (int r = 0; r < 8; ++r)
        xB[(size_t)(m0 + i * 16 + half * 8 + r) * SDIM + (n0 + j * 16 + col)] =
            acc[i][j][r];
}

// ---------------------------------------------------------------------------
// Kernel 2: sequential recurrence  h_t = tanh(h_{t-1} @ A + xB_t)
// One workgroup, 512 thr = 16 waves; wave w owns output columns [16w,16w+16).
// A^T DMA'd into LDS by the Tensor Data Mover (128 KB; 320 KB/WGP makes this
// possible); each wave's eight 32x16 B-fragments of A stay register-resident
// for all 2048 steps. h double-buffered in LDS (batch 8 padded to 16).
// ---------------------------------------------------------------------------
__global__ void __launch_bounds__(512) k_recurrence(const __bf16* __restrict__ AT_bf,
                                                    const float* __restrict__ xB,
                                                    __bf16* __restrict__ hs) {
  extern __shared__ __bf16 smem[];
  __bf16* AsT = smem;                     // 256*256 bf16 = 128 KB at LDS offset 0
  __bf16* hbase = smem + SDIM * SDIM;     // 2 * 16*256 bf16 (ping-pong)

  const int tid = threadIdx.x;
  const int lane = tid & 31, wave = tid >> 5;
  const int half = lane >> 4, col = lane & 15;
  const int n0 = wave * 16;

#if defined(__has_builtin) && __has_builtin(__builtin_amdgcn_tensor_load_to_lds)
  if (wave == 0) {
    // D# per CDNA5 ISA §8: 2-D tensor 256x256, data_size=2B, tile == tensor.
    const unsigned long long ga = (unsigned long long)(const void*)AT_bf;
    v4u g0 = {0u, 0u, 0u, 0u};
    g0[0] = 1u;                                          // count=1, user descriptor
    g0[1] = 0u;                                          // lds_addr = 0 (AsT)
    g0[2] = (unsigned)(ga & 0xFFFFFFFFull);              // global_addr[31:0]
    g0[3] = (unsigned)((ga >> 32) & 0x1FFFFFFull) | (2u << 30);  // addr[56:32], type=2
    v8i g1 = {0, 0, 0, 0, 0, 0, 0, 0};
    g1[0] = (int)(1u << 16);        // data_size=1 (2 bytes)
    g1[1] = (int)(256u << 16);      // tensor_dim0 = 256  (bits [79:48])
    g1[2] = (int)(256u << 16);      // tensor_dim1 = 256  (bits [111:80])
    g1[3] = (int)(256u << 16);      // tile_dim0 = 256    (bits [127:112])
    g1[4] = (int)256;               // tile_dim1 = 256    (bits [143:128])
    g1[5] = (int)256;               // tensor_dim0_stride = 256 (bits [207:160])
    v4i g2 = {0, 0, 0, 0};
    v4i g3 = {0, 0, 0, 0};
    v8i g4 = {0, 0, 0, 0, 0, 0, 0, 0};
    __builtin_amdgcn_tensor_load_to_lds(g0, g1, g2, g3, g4, 0);
    __builtin_amdgcn_s_wait_tensorcnt(0);
  }
#else
  for (int i = tid; i < SDIM * SDIM; i += 512) AsT[i] = AT_bf[i];
#endif
  for (int i = tid; i < 2 * 16 * SDIM; i += 512) hbase[i] = (__bf16)0.0f;
  __syncthreads();

  v16bf bfrag[8];
#pragma unroll
  for (int kt = 0; kt < 8; ++kt)
    bfrag[kt] = load_bt_bf16(AsT, SDIM, kt * 32, n0 + col, lane);

  int cur = 0;
  for (int t = 0; t < LEN; ++t) {
    const __bf16* hc = hbase + cur * (16 * SDIM);
    __bf16* hn = hbase + (cur ^ 1) * (16 * SDIM);

    v8f acc = {};
#pragma unroll
    for (int kt = 0; kt < 8; ++kt) {
      v16bf af = load_a_bf16(hc, SDIM, col, kt * 32, lane);
      acc = wmma_bf16(af, bfrag[kt], acc);
    }
    if (half == 0) {  // rows M=0..7 are real batches; add input projection
#pragma unroll
      for (int r = 0; r < 8; ++r)
        acc[r] += xB[((size_t)r * LEN + t) * SDIM + n0 + col];
    }
    float hv[8];
#pragma unroll
    for (int r = 0; r < 8; ++r) hv[r] = tanhf(acc[r]);  // rows 8..15 stay 0

#pragma unroll
    for (int r = 0; r < 8; ++r) {
      hn[(half * 8 + r) * SDIM + n0 + col] = (__bf16)hv[r];
      if (half == 0)
        hs[((size_t)r * LEN + t) * SDIM + n0 + col] = (__bf16)hv[r];
    }
    __syncthreads();
    cur ^= 1;
  }
}

// ---------------------------------------------------------------------------
// Kernel 3: u = (hs @ C) * sigmoid(dt) -> bf16 (16384x1024), B from CT[d][s]
// ---------------------------------------------------------------------------
__global__ void __launch_bounds__(256) k_gemm_yc(const __bf16* __restrict__ Hs,
                                                 const __bf16* __restrict__ CT,
                                                 const float* __restrict__ dt,
                                                 __bf16* __restrict__ U) {
  const int lane = threadIdx.x & 31;
  const int wave = threadIdx.x >> 5;
  const int m0 = blockIdx.x * 128 + (wave >> 1) * 32;
  const int n0 = blockIdx.y * 64 + (wave & 1) * 32;
  const int half = lane >> 4, col = lane & 15;

  v8f acc[2][2] = {{}, {}};
  for (int k0 = 0; k0 < SDIM; k0 += 32) {
    v16bf a0 = load_a_bf16(Hs, SDIM, m0 + col, k0, lane);
    v16bf a1 = load_a_bf16(Hs, SDIM, m0 + 16 + col, k0, lane);
    v16bf b0 = load_bt_bf16(CT, SDIM, k0, n0 + col, lane);
    v16bf b1 = load_bt_bf16(CT, SDIM, k0, n0 + 16 + col, lane);
    acc[0][0] = wmma_bf16(a0, b0, acc[0][0]);
    acc[0][1] = wmma_bf16(a0, b1, acc[0][1]);
    acc[1][0] = wmma_bf16(a1, b0, acc[1][0]);
    acc[1][1] = wmma_bf16(a1, b1, acc[1][1]);
  }
#pragma unroll
  for (int j = 0; j < 2; ++j) {
    const int n = n0 + j * 16 + col;
    const float g = 1.0f / (1.0f + expf(-dt[n]));  // sigmoid gate, fixed per column
#pragma unroll
    for (int i = 0; i < 2; ++i)
#pragma unroll
      for (int r = 0; r < 8; ++r)
        U[(size_t)(m0 + i * 16 + half * 8 + r) * DIM + n] =
            (__bf16)(acc[i][j][r] * g);
  }
}

// ---------------------------------------------------------------------------
// Kernel 4: causal conv1d as 3 accumulated GEMMs, B from wtT[kk][dout][din]:
//   out[m][dout] = conv_b[dout] + sum_kk sum_din U[b, l-2+kk, din] * w[dout,din,kk]
// ---------------------------------------------------------------------------
__device__ __forceinline__ v16bf load_a_shift(const __bf16* __restrict__ U, int row,
                                              int shift, int k0, int lane) {
  const int b = row >> 11;
  const int l = (row & (LEN - 1)) + shift;
  if ((unsigned)l < (unsigned)LEN)
    return load_a_bf16(U, DIM, (b << 11) + l, k0, lane);
  return zero_a16();
}

__global__ void __launch_bounds__(256) k_gemm_conv(const __bf16* __restrict__ U,
                                                   const __bf16* __restrict__ WtT,
                                                   const float* __restrict__ convb,
                                                   float* __restrict__ out) {
  const int lane = threadIdx.x & 31;
  const int wave = threadIdx.x >> 5;
  const int m0 = blockIdx.x * 128 + (wave >> 1) * 32;
  const int n0 = blockIdx.y * 64 + (wave & 1) * 32;
  const int half = lane >> 4, col = lane & 15;

  v8f acc[2][2] = {{}, {}};
#pragma unroll
  for (int kk = 0; kk < 3; ++kk) {
    const __bf16* W = WtT + (size_t)kk * DIM * DIM;  // [dout][din]
    const int shift = kk - 2;  // causal left-pad of 2
    for (int k0 = 0; k0 < DIM; k0 += 32) {
      __builtin_prefetch(W + (size_t)(n0 + col) * DIM + k0 + 128, 0, 0);
      v16bf a0 = load_a_shift(U, m0 + col, shift, k0, lane);
      v16bf a1 = load_a_shift(U, m0 + 16 + col, shift, k0, lane);
      v16bf b0 = load_bt_bf16(W, DIM, k0, n0 + col, lane);
      v16bf b1 = load_bt_bf16(W, DIM, k0, n0 + 16 + col, lane);
      acc[0][0] = wmma_bf16(a0, b0, acc[0][0]);
      acc[0][1] = wmma_bf16(a0, b1, acc[0][1]);
      acc[1][0] = wmma_bf16(a1, b0, acc[1][0]);
      acc[1][1] = wmma_bf16(a1, b1, acc[1][1]);
    }
  }
#pragma unroll
  for (int j = 0; j < 2; ++j) {
    const int n = n0 + j * 16 + col;
    const float bias = convb[n];
#pragma unroll
    for (int i = 0; i < 2; ++i)
#pragma unroll
      for (int r = 0; r < 8; ++r)
        out[(size_t)(m0 + i * 16 + half * 8 + r) * DIM + n] = acc[i][j][r] + bias;
  }
}

// ---------------------------------------------------------------------------
// Host-side launch
// ---------------------------------------------------------------------------
extern "C" void kernel_launch(void* const* d_in, const int* in_sizes, int n_in,
                              void* d_out, int out_size, void* d_ws, size_t ws_size,
                              hipStream_t stream) {
  const float* x      = (const float*)d_in[0];
  const float* A      = (const float*)d_in[1];
  const float* B_mat  = (const float*)d_in[2];
  const float* C      = (const float*)d_in[3];
  const float* dt     = (const float*)d_in[4];
  const float* conv_w = (const float*)d_in[5];
  const float* conv_b = (const float*)d_in[6];
  float* out = (float*)d_out;

  // Workspace layout (bytes), ~95 MB total
  char* ws = (char*)d_ws;
  size_t off = 0;
  __bf16* x_bf   = (__bf16*)(ws + off); off += (size_t)ROWS * DIM * 2;       // 32 MB
  __bf16* BmT_bf = (__bf16*)(ws + off); off += (size_t)DIM * SDIM * 2;       // 512 KB
  __bf16* AT_bf  = (__bf16*)(ws + off); off += (size_t)SDIM * SDIM * 2;      // 128 KB
  __bf16* CT_bf  = (__bf16*)(ws + off); off += (size_t)SDIM * DIM * 2;       // 512 KB
  __bf16* wtT_bf = (__bf16*)(ws + off); off += (size_t)3 * DIM * DIM * 2;    // 6 MB
  float*  xB     = (float*) (ws + off); off += (size_t)ROWS * SDIM * 4;      // 16 MB
  __bf16* hs     = (__bf16*)(ws + off); off += (size_t)ROWS * SDIM * 2;      // 8 MB
  __bf16* U      = (__bf16*)(ws + off); off += (size_t)ROWS * DIM * 2;       // 32 MB

  // 0) precision conversion + B-operand transposes
  k_convert<<<4096, 256, 0, stream>>>(x, A, B_mat, C, conv_w,
                                      x_bf, AT_bf, BmT_bf, CT_bf, wtT_bf);

  // 1) input projection GEMM
  k_gemm_xb<<<dim3(ROWS / 128, SDIM / 64), 256, 0, stream>>>(x_bf, BmT_bf, xB);

  // 2) sequential scan (single WGP; A^T TDM'd into LDS, slabs in registers)
  const size_t smem_bytes = (size_t)(SDIM * SDIM + 2 * 16 * SDIM) * sizeof(__bf16);
  k_recurrence<<<1, 512, smem_bytes, stream>>>(AT_bf, xB, hs);

  // 3) output projection + sigmoid(dt) gate
  k_gemm_yc<<<dim3(ROWS / 128, DIM / 64), 256, 0, stream>>>(hs, CT_bf, dt, U);

  // 4) causal conv1d as 3 accumulated GEMMs + bias
  k_gemm_conv<<<dim3(ROWS / 128, DIM / 64), 256, 0, stream>>>(U, wtT_bf, conv_b, out);

  (void)in_sizes; (void)n_in; (void)out_size; (void)ws_size;
}
